// Attention_2576980377638
// MI455X (gfx1250) — matrix operand order repaired and checked by reading, not compiled, not run on hardware
//
#include <hip/hip_runtime.h>
#include <hip/hip_bf16.h>

// ---------------------------------------------------------------------------
// Problem constants (match reference)
// ---------------------------------------------------------------------------
constexpr int kB = 1024;   // batch
constexpr int kT = 64;     // encoder length
constexpr int kD = 512;    // input size
constexpr int kH = 512;    // hidden
constexpr int kC = 96;     // classes
constexpr int kS = 21;     // decode steps
constexpr int kDC = kD + kC;   // 608, Wih row length
constexpr int kKcat = kD + kH; // 1024, concat GEMM K

typedef __bf16 bf16_t;
typedef bf16_t v2bf  __attribute__((ext_vector_type(2)));
typedef bf16_t v8bf  __attribute__((ext_vector_type(8)));
typedef bf16_t v16bf __attribute__((ext_vector_type(16)));
typedef float  v4f   __attribute__((ext_vector_type(4)));
typedef float  v8f   __attribute__((ext_vector_type(8)));

__device__ __forceinline__ float sigmoidf_(float x) { return 1.0f / (1.0f + __expf(-x)); }

__device__ __forceinline__ v16bf join16(v8bf lo, v8bf hi) {
  return __builtin_shufflevector(lo, hi, 0, 1, 2, 3, 4, 5, 6, 7, 8, 9, 10, 11,
                                 12, 13, 14, 15);
}

// ---------------------------------------------------------------------------
// Generic TN GEMM with bf16 WMMA:  C[m,n] = sum_k A[m,k] * W[n,k] (+ bias[n])
// A: [M, lda] bf16 row-major (K contiguous) ; W: [N, ldb] bf16 row-major.
// One wave computes a 16 x (NT*16) strip; 8 waves/block -> 128 rows/block.
// Fragment layouts per CDNA5 ISA 7.12.2 (wave32):
//   A (16x32 bf16): lane L holds row m=L%16, K-chunks [k0..k0+7],[k0+16..k0+23],
//                   k0 = (L<16 ? 0 : 8).
//   B (32x16 bf16): lane L holds col n=L%16, K-chunks [b0..b0+7],[b0+16..b0+23], b0 = (L<16?0:8).
//   D (16x16 f32):  vgpr j -> row j + 8*(L/16), col L%16.
// k-loop is ping-pong double-buffered with a manual unroll-by-2 so each
// buffer is re-defined directly by its loads (no register rotation copies,
// no WAR-hazard NOPs); K must be a multiple of 64 (all call sites: 512/1024).
// ---------------------------------------------------------------------------
template <int NT, bool OUT_BF16, bool ADD_BIAS>
__global__ __launch_bounds__(256) void gemm_tn_wmma(
    const bf16_t* __restrict__ A, int lda,
    const bf16_t* __restrict__ W, int ldb,
    void* __restrict__ Cout, int ldc,
    const float* __restrict__ bias, int K) {
  const int lane  = threadIdx.x & 31;
  const int wave  = threadIdx.x >> 5;
  const int mtile = blockIdx.y * 128 + wave * 16;
  const int ntile = blockIdx.x * (NT * 16);

  v8f acc[NT] = {};

  const int ak = (lane >> 4) << 3;  // 0 or 8
  const int bk = (lane >> 4) << 3;  // 0 or 8

  const bf16_t* __restrict__ Aptr =
      A + (size_t)(mtile + (lane & 15)) * lda + ak;
  const bf16_t* Bptr[NT];
#pragma unroll
  for (int t = 0; t < NT; ++t)
    Bptr[t] = W + (size_t)(ntile + t * 16 + (lane & 15)) * ldb + bk;

  auto loadA = [&](int kk) -> v16bf {
    return join16(*(const v8bf*)(Aptr + kk), *(const v8bf*)(Aptr + kk + 16));
  };
  auto loadB = [&](int t, int kk) -> v16bf {
    return join16(*(const v8bf*)(Bptr[t] + kk),
                  *(const v8bf*)(Bptr[t] + kk + 16));
  };

  const int niter = K >> 5;  // even (K % 64 == 0)

  // prologue: buffer0 <- k-step 0, buffer1 <- k-step 1
  v16bf a0 = loadA(0);
  v16bf a1 = loadA(32);
  v16bf b0[NT], b1[NT];
#pragma unroll
  for (int t = 0; t < NT; ++t) {
    b0[t] = loadB(t, 0);
    b1[t] = loadB(t, 32);
  }

  for (int i = 0; i < niter; i += 2) {
    // ---- k-step i: consume buffer0, then refill it for step i+2 ----
#pragma unroll
    for (int t = 0; t < NT; ++t)
      acc[t] = __builtin_amdgcn_wmma_f32_16x16x32_bf16(
          false, a0, false, b0[t], (short)0, acc[t], false, false);
    if (i + 2 < niter) {
      const int kn = (i + 2) << 5;
      a0 = loadA(kn);
#pragma unroll
      for (int t = 0; t < NT; ++t) b0[t] = loadB(t, kn);
    }
    // ---- k-step i+1: consume buffer1, then refill it for step i+3 ----
#pragma unroll
    for (int t = 0; t < NT; ++t)
      acc[t] = __builtin_amdgcn_wmma_f32_16x16x32_bf16(
          false, a1, false, b1[t], (short)0, acc[t], false, false);
    if (i + 3 < niter) {
      const int kn = (i + 3) << 5;
      a1 = loadA(kn);
#pragma unroll
      for (int t = 0; t < NT; ++t) b1[t] = loadB(t, kn);
    }
  }

  const int mrow = (lane >> 4) * 8;
  const int ncol = lane & 15;
#pragma unroll
  for (int t = 0; t < NT; ++t) {
    const int n = ntile + t * 16 + ncol;
    const float badd = ADD_BIAS ? bias[n] : 0.0f;
#pragma unroll
    for (int j = 0; j < 8; ++j) {
      const int m = mtile + mrow + j;
      const float v = acc[t][j] + badd;
      if (OUT_BF16)
        ((bf16_t*)Cout)[(size_t)m * ldc + n] = (bf16_t)v;
      else
        ((float*)Cout)[(size_t)m * ldc + n] = v;
    }
  }
}

// ---------------------------------------------------------------------------
// Prep kernels
// ---------------------------------------------------------------------------
__global__ __launch_bounds__(256) void cast_f32_bf16(const float* __restrict__ in,
                                                     bf16_t* __restrict__ out,
                                                     int n) {
  int i = 4 * (blockIdx.x * 256 + threadIdx.x);
#pragma unroll
  for (int k = 0; k < 4; ++k)
    if (i + k < n) out[i + k] = (bf16_t)in[i + k];
}

// Wcat[j, 0:512] = Wih[j, 0:512] ; Wcat[j, 512:1024] = Whh[j, :]
__global__ __launch_bounds__(256) void pack_wcat(const float* __restrict__ Wih,
                                                 const float* __restrict__ Whh,
                                                 bf16_t* __restrict__ out) {
  int idx = blockIdx.x * 256 + threadIdx.x;  // over 2048*1024
  int j = idx >> 10;
  int k = idx & 1023;
  float v = (k < kD) ? Wih[(size_t)j * kDC + k] : Whh[(size_t)j * kH + (k - kD)];
  out[idx] = (bf16_t)v;
}

// tokcol[c, j] = bih[j] + bhh[j] + Wih[j, D + c]   (folds one-hot + biases)
__global__ __launch_bounds__(256) void make_tokcol(const float* __restrict__ Wih,
                                                   const float* __restrict__ bih,
                                                   const float* __restrict__ bhh,
                                                   float* __restrict__ out) {
  int idx = blockIdx.x * 256 + threadIdx.x;  // over kC * 4H
  if (idx >= kC * 4 * kH) return;
  int c = idx >> 11;        // /2048
  int j = idx & 2047;
  out[idx] = bih[j] + bhh[j] + Wih[(size_t)j * kDC + kD + c];
}

// zero LSTM cell state and the h half of the concat GEMM input
__global__ __launch_bounds__(256) void init_state(float* __restrict__ cst,
                                                  bf16_t* __restrict__ xcat) {
  int idx = blockIdx.x * 256 + threadIdx.x;  // over B*H
  int b = idx >> 9;
  int h = idx & 511;
  cst[idx] = 0.0f;
  xcat[(size_t)b * kKcat + kD + h] = (bf16_t)0.0f;
}

// ---------------------------------------------------------------------------
// Fused attention step: one block per batch row b.
//   e[t] = sum_h tanh(Hp[b,t,h] + ph[b,h]) * Ws[h]   (wave per t, 8 waves)
//   alpha = softmax_t(e) ; ctx[b,d] = sum_t alpha[t]*batch_H[b,t,d]
// ctx written as bf16 into xcat[b, 0:512] (the gates-GEMM A operand).
// ---------------------------------------------------------------------------
__global__ __launch_bounds__(256) void attention_step(
    const bf16_t* __restrict__ Hp16, const bf16_t* __restrict__ bh16,
    const float* __restrict__ ph, const float* __restrict__ Ws,
    bf16_t* __restrict__ xcat) {
  const int b = blockIdx.x;
  const int tid = threadIdx.x;
  const int lane = tid & 31;
  const int wave = tid >> 5;

  __shared__ float e_s[kT];
  __shared__ float a_s[kT];

  // each lane owns h = [lane*16, lane*16+16)
  const int h0 = lane * 16;
  float wsr[16], phr[16];
  {
    const float* __restrict__ phb = ph + (size_t)b * kH + h0;
    const float* __restrict__ wsb = Ws + h0;
#pragma unroll
    for (int q = 0; q < 4; ++q) {
      v4f w = *(const v4f*)(wsb + 4 * q);
      v4f p = *(const v4f*)(phb + 4 * q);
#pragma unroll
      for (int r = 0; r < 4; ++r) {
        wsr[4 * q + r] = w[r];
        phr[4 * q + r] = p[r];
      }
    }
  }

  for (int t = wave; t < kT; t += 8) {
    const bf16_t* __restrict__ hp = Hp16 + ((size_t)b * kT + t) * kH + h0;
    __builtin_prefetch(hp + 8 * kH, 0, 1);  // next t this wave will score
    v8bf p0 = *(const v8bf*)(hp);
    v8bf p1 = *(const v8bf*)(hp + 8);
    float sum = 0.0f;
#pragma unroll
    for (int i = 0; i < 8; ++i) sum += tanhf((float)p0[i] + phr[i]) * wsr[i];
#pragma unroll
    for (int i = 0; i < 8; ++i)
      sum += tanhf((float)p1[i] + phr[8 + i]) * wsr[8 + i];
#pragma unroll
    for (int off = 16; off > 0; off >>= 1) sum += __shfl_down(sum, off, 32);
    if (lane == 0) e_s[t] = sum;
  }
  __syncthreads();

  // softmax over T=64 (cheap, done redundantly per thread from LDS)
  float mx = -3.4e38f;
  for (int t = 0; t < kT; ++t) mx = fmaxf(mx, e_s[t]);
  float den = 0.0f;
  for (int t = 0; t < kT; ++t) den += __expf(e_s[t] - mx);
  if (tid < kT) a_s[tid] = __expf(e_s[tid] - mx) / den;
  __syncthreads();

  // context: thread owns the bf16 pair d = {2*tid, 2*tid+1}; dword loads
  float c0 = 0.0f, c1 = 0.0f;
  const uint32_t* __restrict__ bp32 =
      (const uint32_t*)(bh16 + (size_t)b * kT * kD);
  for (int t = 0; t < kT; ++t) {
    const float al = a_s[t];
    const uint32_t w = bp32[t * (kD / 2) + tid];
    c0 += al * __uint_as_float(w << 16);           // low bf16
    c1 += al * __uint_as_float(w & 0xffff0000u);   // high bf16
  }
  v2bf r;
  r.x = (bf16_t)c0;
  r.y = (bf16_t)c1;
  *(v2bf*)(xcat + (size_t)b * kKcat + 2 * tid) = r;
}

// ---------------------------------------------------------------------------
// Pointwise LSTM update. gates layout [B, 4H] = [i | f | g | o].
// Adds tokcol (biases + one-hot column), updates c, writes h (bf16) into the
// concat GEMM input and into the hidden history.
// ---------------------------------------------------------------------------
__global__ __launch_bounds__(256) void lstm_step(
    const float* __restrict__ gates, const float* __restrict__ tokcol,
    const int* __restrict__ text, float* __restrict__ cst,
    bf16_t* __restrict__ xcat, bf16_t* __restrict__ hist, int s) {
  int idx = blockIdx.x * 256 + threadIdx.x;  // over B*H
  int b = idx >> 9;
  int h = idx & 511;
  int tok = text[b * kS + s];
  const float* __restrict__ tc = tokcol + (size_t)tok * (4 * kH);
  size_t gb = (size_t)b * (4 * kH);

  float gi = gates[gb + h]          + tc[h];
  float gf = gates[gb + kH + h]     + tc[kH + h];
  float gg = gates[gb + 2 * kH + h] + tc[2 * kH + h];
  float go = gates[gb + 3 * kH + h] + tc[3 * kH + h];

  float c2 = sigmoidf_(gf) * cst[idx] + sigmoidf_(gi) * tanhf(gg);
  float h2 = sigmoidf_(go) * tanhf(c2);

  cst[idx] = c2;
  xcat[(size_t)b * kKcat + kD + h] = (bf16_t)h2;
  hist[((size_t)b * kS + s) * kH + h] = (bf16_t)h2;
}

// ---------------------------------------------------------------------------
// Host orchestration
// ---------------------------------------------------------------------------
extern "C" void kernel_launch(void* const* d_in, const int* in_sizes, int n_in,
                              void* d_out, int out_size, void* d_ws,
                              size_t ws_size, hipStream_t stream) {
  const float* batch_H = (const float*)d_in[0];
  const int*   text    = (const int*)d_in[1];
  const float* Wi      = (const float*)d_in[2];
  const float* Wh      = (const float*)d_in[3];
  const float* bh      = (const float*)d_in[4];
  const float* Ws      = (const float*)d_in[5];
  const float* Wih     = (const float*)d_in[6];
  const float* Whh     = (const float*)d_in[7];
  const float* bih     = (const float*)d_in[8];
  const float* bhh     = (const float*)d_in[9];
  const float* Wg      = (const float*)d_in[10];
  const float* bg      = (const float*)d_in[11];
  float* out = (float*)d_out;

  // ---- workspace carve (256B aligned) ----
  char* base = (char*)d_ws;
  size_t off = 0;
  auto take = [&](size_t bytes) -> void* {
    void* r = base + off;
    off = (off + bytes + 255) & ~(size_t)255;
    return r;
  };
  bf16_t* bh16   = (bf16_t*)take((size_t)kB * kT * kD * 2);  // 67 MB
  bf16_t* Hp16   = (bf16_t*)take((size_t)kB * kT * kH * 2);  // 67 MB
  bf16_t* Wi16   = (bf16_t*)take((size_t)kH * kD * 2);
  bf16_t* Wh16   = (bf16_t*)take((size_t)kH * kH * 2);
  bf16_t* Wcat   = (bf16_t*)take((size_t)4 * kH * kKcat * 2);
  bf16_t* Wg16   = (bf16_t*)take((size_t)kC * kH * 2);
  float*  tokcol = (float*) take((size_t)kC * 4 * kH * 4);
  bf16_t* xcat   = (bf16_t*)take((size_t)kB * kKcat * 2);
  float*  cstate = (float*) take((size_t)kB * kH * 4);
  float*  ph     = (float*) take((size_t)kB * kH * 4);
  float*  gates  = (float*) take((size_t)kB * 4 * kH * 4);
  bf16_t* hist   = (bf16_t*)take((size_t)kB * kS * kH * 2);
  (void)ws_size; (void)n_in; (void)in_sizes; (void)out_size;

  // ---- one-time prep ----
  {
    int n = kB * kT * kD;
    cast_f32_bf16<<<n / 1024, 256, 0, stream>>>(batch_H, bh16, n);
    n = kH * kD;
    cast_f32_bf16<<<n / 1024, 256, 0, stream>>>(Wi, Wi16, n);
    n = kH * kH;
    cast_f32_bf16<<<n / 1024, 256, 0, stream>>>(Wh, Wh16, n);
    n = kC * kH;
    cast_f32_bf16<<<n / 1024, 256, 0, stream>>>(Wg, Wg16, n);
    pack_wcat<<<(4 * kH * kKcat) / 256, 256, 0, stream>>>(Wih, Whh, Wcat);
    make_tokcol<<<(kC * 4 * kH + 255) / 256, 256, 0, stream>>>(Wih, bih, bhh,
                                                               tokcol);
    init_state<<<(kB * kH) / 256, 256, 0, stream>>>(cstate, xcat);
  }

  // ---- Hp = batch_H @ Wi.T  (bf16 out: re-read 21x, fits L2) ----
  gemm_tn_wmma<4, true, false>
      <<<dim3(kH / 64, (kB * kT) / 128), 256, 0, stream>>>(
          bh16, kD, Wi16, kD, Hp16, kH, nullptr, kD);

  // ---- decode steps ----
  for (int s = 0; s < kS; ++s) {
    // ph = h @ Wh.T + bh   (A = h half of xcat)
    gemm_tn_wmma<4, false, true>
        <<<dim3(kH / 64, kB / 128), 256, 0, stream>>>(
            xcat + kD, kKcat, Wh16, kH, ph, kH, bh, kH);

    // attention -> ctx (bf16 into xcat[:, 0:512])
    attention_step<<<kB, 256, 0, stream>>>(Hp16, bh16, ph, Ws, xcat);

    // gates = [ctx|h] @ [Wih[:,:D]|Whh].T
    gemm_tn_wmma<4, false, false>
        <<<dim3((4 * kH) / 64, kB / 128), 256, 0, stream>>>(
            xcat, kKcat, Wcat, kKcat, gates, 4 * kH, nullptr, kKcat);

    // pointwise LSTM
    lstm_step<<<(kB * kH) / 256, 256, 0, stream>>>(gates, tokcol, text, cstate,
                                                   xcat, hist, s);
  }

  // ---- probs = hist @ Wg.T + bg ----
  gemm_tn_wmma<2, false, true>
      <<<dim3(kC / 32, (kB * kS) / 128), 256, 0, stream>>>(
          hist, kH, Wg16, kH, out, kC, bg, kH);
}
